// TransformerEncoder_23390391894408
// MI455X (gfx1250) — matrix operand order repaired and checked by reading, not compiled, not run on hardware
//
#include <hip/hip_runtime.h>

typedef __attribute__((ext_vector_type(16))) _Float16 v16h;
typedef __attribute__((ext_vector_type(8)))  float    v8f;
typedef __attribute__((ext_vector_type(4)))  int      v4i;

#define NT     25600   // B*S tokens
#define NPAD   25728   // padded rows (>= 127*200 + 255 + 1)
#define DMODEL 512
#define NHEAD  8
#define HDIM   64
#define FFDIM  2048
#define NLAYER 6
#define SEQ    200
#define SPAD   224     // padded seq (7 chunks of 32 keys)
#define BATCH  128

#if defined(__has_builtin)
#if __has_builtin(__builtin_amdgcn_global_load_async_to_lds_b128) && \
    __has_builtin(__builtin_amdgcn_s_wait_asynccnt)
#define USE_ASYNC_LDS 1
#endif
#endif

// 16-byte global->LDS copy: async DMA path on CDNA5, sync fallback otherwise.
__device__ __forceinline__ void cp_async16(_Float16* lds, const _Float16* g) {
#ifdef USE_ASYNC_LDS
  __builtin_amdgcn_global_load_async_to_lds_b128(
      (__attribute__((address_space(1))) v4i*)(_Float16*)g,
      (__attribute__((address_space(3))) v4i*)lds, 0, 0);
#else
  *reinterpret_cast<float4*>(lds) = *reinterpret_cast<const float4*>(g);
#endif
}
__device__ __forceinline__ void staging_wait() {
#ifdef USE_ASYNC_LDS
  __builtin_amdgcn_s_wait_asynccnt(0);
#endif
}

__device__ __forceinline__ v8f zero_v8f() {
  v8f z;
#pragma unroll
  for (int i = 0; i < 8; ++i) z[i] = 0.0f;
  return z;
}

__device__ __forceinline__ v8f wmma_f16(v16h a, v16h b, v8f c) {
  // (neg_a, A, neg_b, B, c_mod, C, reuse_a, reuse_b)
  return __builtin_amdgcn_wmma_f32_16x16x32_f16(false, a, false, b, (short)0, c,
                                                false, false);
}

// A fragment 16x32 f16 from LDS tile (row-major, lda in halves).
// lane<16: row=lane, K 0..7 in v0..3, K 16..23 in v4..7
// lane>=16: row=lane-16, K 8..15 and 24..31.
__device__ __forceinline__ v16h frag_a(const _Float16* base, int lda) {
  const int lane = threadIdx.x & 31;
  const _Float16* p = base + (lane & 15) * lda + ((lane >> 4) << 3);
  v16h a;
  float4* av = reinterpret_cast<float4*>(&a);
  av[0] = *reinterpret_cast<const float4*>(p);
  av[1] = *reinterpret_cast<const float4*>(p + 16);
  return a;
}

// B fragment 32x16 f16 from a column-staged tile bt[n][k] (ldb in halves).
// lane<16: n=lane, K 0..7 and 16..23 ; lane>=16: n=lane-16, K 8..15 and 24..31.
__device__ __forceinline__ v16h frag_b(const _Float16* base, int ldb) {
  const int lane = threadIdx.x & 31;
  const _Float16* p = base + (lane & 15) * ldb + ((lane >> 4) << 3);
  v16h b;
  float4* bv = reinterpret_cast<float4*>(&b);
  bv[0] = *reinterpret_cast<const float4*>(p);
  bv[1] = *reinterpret_cast<const float4*>(p + 16);
  return b;
}

__device__ __forceinline__ float wave_max16(float v) {
  v = fmaxf(v, __shfl_xor(v, 1, 32));
  v = fmaxf(v, __shfl_xor(v, 2, 32));
  v = fmaxf(v, __shfl_xor(v, 4, 32));
  v = fmaxf(v, __shfl_xor(v, 8, 32));
  return v;
}
__device__ __forceinline__ float wave_sum16(float v) {
  v += __shfl_xor(v, 1, 32);
  v += __shfl_xor(v, 2, 32);
  v += __shfl_xor(v, 4, 32);
  v += __shfl_xor(v, 8, 32);
  return v;
}

// ------------------------------------------------ weight prep (once/launch)
// W[K][N] f32 -> Wt[N][K] f16   (N = 1<<nShift)
__global__ __launch_bounds__(256) void transposeW_kernel(
    const float* __restrict__ W, _Float16* __restrict__ Wt, int K, int nShift) {
  const int idx = blockIdx.x * 256 + threadIdx.x;
  if (idx >= (K << nShift)) return;
  const int kk = idx >> nShift;
  const int nn = idx & ((1 << nShift) - 1);
  Wt[(size_t)nn * K + kk] = (_Float16)W[idx];
}

__global__ __launch_bounds__(256) void zero_h_kernel(_Float16* __restrict__ p,
                                                     int n) {
  const int idx = blockIdx.x * 256 + threadIdx.x;
  if (idx < n) p[idx] = (_Float16)0.0f;
}

// ---------------------------------------------------------------- embedding
__global__ __launch_bounds__(256) void embed_kernel(
    const float* __restrict__ traj, const float* __restrict__ We,
    const float* __restrict__ be, const float* __restrict__ pe,
    float* __restrict__ x32, _Float16* __restrict__ x16) {
  const int idx = blockIdx.x * 256 + threadIdx.x;
  if (idx >= NT * DMODEL) return;
  const int n = idx >> 9;
  const int d = idx & 511;
  const float t0 = traj[n * 2 + 0];
  const float t1 = traj[n * 2 + 1];
  const float v = t0 * We[d] + t1 * We[DMODEL + d] + be[d] + pe[SEQ * DMODEL + d];
  x32[idx] = v;
  x16[idx] = (_Float16)v;
}

// ---------------------------------------------------------------- WMMA GEMM
// out = act(A[f16, MxK] * Wt[f16, NxK]^T + bias)   (128x128 tile / WG)
// OUTVT: write f16 output transposed per-batch: vt[b][col][s]  (SPAD stride)
#define LDS_A 40
#define LDS_B 40

template <bool RELU, bool OUT32, bool OUT16, bool OUTVT>
__global__ __launch_bounds__(256) void gemm_kernel(
    const _Float16* __restrict__ A, const _Float16* __restrict__ Wt,
    const float* __restrict__ bias, float* __restrict__ out32,
    _Float16* __restrict__ out16, int M, int N, int K) {
  __shared__ _Float16 sA[128 * LDS_A];
  __shared__ _Float16 sB[128 * LDS_B];  // [n][k]
  const int tid  = threadIdx.x;
  const int lane = tid & 31;
  const int wave = tid >> 5;
  const int wm = wave >> 1;  // 0..3  -> 32 rows each
  const int wn = wave & 1;   // 0..1  -> 64 cols each
  const int m0 = blockIdx.y * 128;
  const int n0 = blockIdx.x * 128;

  v8f acc[2][4];
#pragma unroll
  for (int i = 0; i < 2; ++i)
#pragma unroll
    for (int j = 0; j < 4; ++j) acc[i][j] = zero_v8f();

  const int sRow = tid >> 1;        // 0..127
  const int sCol = (tid & 1) << 4;  // 0 / 16

  for (int k0 = 0; k0 < K; k0 += 32) {
    __syncthreads();
    const _Float16* gA = A  + (size_t)(m0 + sRow) * K + (k0 + sCol);
    const _Float16* gB = Wt + (size_t)(n0 + sRow) * K + (k0 + sCol);
    cp_async16(&sA[sRow * LDS_A + sCol], gA);
    cp_async16(&sA[sRow * LDS_A + sCol + 8], gA + 8);
    cp_async16(&sB[sRow * LDS_B + sCol], gB);
    cp_async16(&sB[sRow * LDS_B + sCol + 8], gB + 8);
    if (k0 + 32 < K) {  // prefetch next k-step tiles
      __builtin_prefetch(gA + 32, 0, 3);
      __builtin_prefetch(gB + 32, 0, 3);
    }
    staging_wait();
    __syncthreads();

    v16h af[2], bf[4];
#pragma unroll
    for (int i = 0; i < 2; ++i)
      af[i] = frag_a(&sA[(wm * 32 + i * 16) * LDS_A], LDS_A);
#pragma unroll
    for (int j = 0; j < 4; ++j)
      bf[j] = frag_b(&sB[(wn * 64 + j * 16) * LDS_B], LDS_B);
#pragma unroll
    for (int i = 0; i < 2; ++i)
#pragma unroll
      for (int j = 0; j < 4; ++j) acc[i][j] = wmma_f16(af[i], bf[j], acc[i][j]);
  }

#pragma unroll
  for (int i = 0; i < 2; ++i) {
#pragma unroll
    for (int j = 0; j < 4; ++j) {
      const int col  = n0 + wn * 64 + j * 16 + (lane & 15);
      const float bv = bias[col];
      const int rowb = m0 + wm * 32 + i * 16 + ((lane >> 4) << 3);
#pragma unroll
      for (int r = 0; r < 8; ++r) {
        float v = acc[i][j][r] + bv;
        if (RELU) v = fmaxf(v, 0.0f);
        if (OUT32) out32[(size_t)(rowb + r) * N + col] = v;
        if (OUT16) out16[(size_t)(rowb + r) * N + col] = (_Float16)v;
        if (OUTVT) {
          const int bi = (rowb + r) / SEQ;
          const int si = (rowb + r) % SEQ;
          out16[((size_t)bi * DMODEL + col) * SPAD + si] = (_Float16)v;
        }
      }
    }
  }
}

// ------------------------------------------------------- flash attention
#define LDS_K 72
#define LDS_V 40

__global__ __launch_bounds__(256) void attn_kernel(
    const _Float16* __restrict__ q, const _Float16* __restrict__ k,
    const _Float16* __restrict__ vt, const int* __restrict__ mask,
    _Float16* __restrict__ ctx) {
  __shared__ _Float16 sK[32 * LDS_K];  // [key][hd]
  __shared__ _Float16 sV[64 * LDS_V];  // [hd][key]
  __shared__ _Float16 sP[8][16 * 32];  // per-wave probs tile
  const int tid  = threadIdx.x;
  const int lane = tid & 31;
  const int wave = tid >> 5;
  const int half = lane >> 4;
  const int cl   = lane & 15;
  const int b = blockIdx.z;
  const int h = blockIdx.y;
  const int sq0 = blockIdx.x * 128 + wave * 16;
  const size_t rowQ0 = (size_t)b * SEQ + sq0;  // padded buffer: in bounds

  // Q fragments (16 queries x 64 hd) straight from global
  v16h qa[2];
#pragma unroll
  for (int c = 0; c < 2; ++c) {
    const _Float16* p = q + (rowQ0 + cl) * DMODEL + h * HDIM + c * 32 + half * 8;
    float4* av = reinterpret_cast<float4*>(&qa[c]);
    av[0] = *reinterpret_cast<const float4*>(p);
    av[1] = *reinterpret_cast<const float4*>(p + 16);
  }

  v8f accO[4];
#pragma unroll
  for (int f = 0; f < 4; ++f) accO[f] = zero_v8f();
  float mrow[8], lrow[8];
#pragma unroll
  for (int r = 0; r < 8; ++r) { mrow[r] = -3.0e38f; lrow[r] = 0.0f; }

  const int stKey = tid >> 3;        // 0..31
  const int stHd  = (tid & 7) << 3;  // 0..56
  const int vHd   = tid >> 2;        // 0..63
  const int vKo   = (tid & 3) << 3;  // 0..24

  for (int kc = 0; kc < SPAD; kc += 32) {
    __syncthreads();
    // K rows past SEQ hold padded garbage; their scores are masked below.
    cp_async16(&sK[stKey * LDS_K + stHd],
               k + ((size_t)b * SEQ + kc + stKey) * DMODEL + h * HDIM + stHd);
    // V is pre-transposed [b][col][s] with zeroed pad keys.
    cp_async16(&sV[vHd * LDS_V + vKo],
               vt + ((size_t)b * DMODEL + h * HDIM + vHd) * SPAD + kc + vKo);
    staging_wait();
    __syncthreads();

    // scores: 16q x 32keys via 2x2 WMMAs over hd=64
    v8f sc[2];
#pragma unroll
    for (int nf = 0; nf < 2; ++nf) sc[nf] = zero_v8f();
#pragma unroll
    for (int c = 0; c < 2; ++c)
#pragma unroll
      for (int nf = 0; nf < 2; ++nf) {
        v16h bf = frag_b(&sK[(nf * 16) * LDS_K + c * 32], LDS_K);
        sc[nf] = wmma_f16(qa[c], bf, sc[nf]);
      }

    // scale + mask
#pragma unroll
    for (int nf = 0; nf < 2; ++nf) {
      const int key = kc + nf * 16 + cl;
      const bool ok = (key < SEQ) && (mask[b * SEQ + key] != 0);
#pragma unroll
      for (int r = 0; r < 8; ++r)
        sc[nf][r] = ok ? sc[nf][r] * 0.125f : -3.0e38f;
    }

    // online softmax (rows r+8*half, reduce over 16-lane halves)
    float corr[8];
#pragma unroll
    for (int r = 0; r < 8; ++r) {
      const float mx = wave_max16(fmaxf(sc[0][r], sc[1][r]));
      const float mn = fmaxf(mrow[r], mx);
      corr[r] = __expf(mrow[r] - mn);
      mrow[r] = mn;
    }
#pragma unroll
    for (int r = 0; r < 8; ++r) {
      const float p0 = __expf(sc[0][r] - mrow[r]);
      const float p1 = __expf(sc[1][r] - mrow[r]);
      sP[wave][(r + 8 * half) * 32 + cl] = (_Float16)p0;
      sP[wave][(r + 8 * half) * 32 + 16 + cl] = (_Float16)p1;
      lrow[r] = lrow[r] * corr[r] + wave_sum16(p0 + p1);
    }
#pragma unroll
    for (int f = 0; f < 4; ++f)
#pragma unroll
      for (int r = 0; r < 8; ++r) accO[f][r] *= corr[r];

    __syncthreads();  // sP visibility + uniform barrier
    v16h pa = frag_a(&sP[wave][0], 32);
#pragma unroll
    for (int f = 0; f < 4; ++f) {
      v16h bf = frag_b(&sV[(f * 16) * LDS_V], LDS_V);
      accO[f] = wmma_f16(pa, bf, accO[f]);
    }
  }

#pragma unroll
  for (int f = 0; f < 4; ++f)
#pragma unroll
    for (int r = 0; r < 8; ++r) {
      const int sq = sq0 + r + 8 * half;
      if (sq < SEQ)
        ctx[((size_t)b * SEQ + sq) * DMODEL + h * HDIM + f * 16 + cl] =
            (_Float16)(accO[f][r] / lrow[r]);
    }
}

// ------------------------------------------------- residual + layernorm
template <bool HAS_RES, bool OUT32, bool OUT16>
__global__ __launch_bounds__(256) void ln_kernel(
    const float* __restrict__ x, const float* __restrict__ res,
    const float* __restrict__ g, const float* __restrict__ bb,
    float* __restrict__ out32, _Float16* __restrict__ out16, int nrows) {
  const int lane = threadIdx.x & 31;
  const int wave = threadIdx.x >> 5;
  const int row  = blockIdx.x * 8 + wave;
  if (row >= nrows) return;
  const float* px = x + (size_t)row * DMODEL;
  const float* pr = res + (size_t)row * DMODEL;
  float vals[16];
  float sum = 0.f;
#pragma unroll
  for (int j = 0; j < 16; ++j) {
    float t = px[lane + 32 * j];
    if (HAS_RES) t += pr[lane + 32 * j];
    vals[j] = t;
    sum += t;
  }
  sum += __shfl_xor(sum, 16, 32);
  sum = wave_sum16(sum);
  const float mean = sum * (1.0f / DMODEL);
  float vs = 0.f;
#pragma unroll
  for (int j = 0; j < 16; ++j) {
    const float d = vals[j] - mean;
    vs += d * d;
  }
  vs += __shfl_xor(vs, 16, 32);
  vs = wave_sum16(vs);
  const float rstd = rsqrtf(vs * (1.0f / DMODEL) + 1e-5f);
#pragma unroll
  for (int j = 0; j < 16; ++j) {
    const int c = lane + 32 * j;
    const float o = (vals[j] - mean) * rstd * g[c] + bb[c];
    if (OUT32) out32[(size_t)row * DMODEL + c] = o;
    if (OUT16) out16[(size_t)row * DMODEL + c] = (_Float16)o;
  }
}

// ----------------------------------------------------------------- driver
extern "C" void kernel_launch(void* const* d_in, const int* in_sizes, int n_in,
                              void* d_out, int out_size, void* d_ws,
                              size_t ws_size, hipStream_t stream) {
  const float* traj  = (const float*)d_in[0];
  const int*   mask  = (const int*)d_in[1];
  const float* pe    = (const float*)d_in[2];
  const float* W_emb = (const float*)d_in[3];
  const float* b_emb = (const float*)d_in[4];
  const float* Wq = (const float*)d_in[5];   const float* bq = (const float*)d_in[6];
  const float* Wk = (const float*)d_in[7];   const float* bk = (const float*)d_in[8];
  const float* Wv = (const float*)d_in[9];   const float* bv = (const float*)d_in[10];
  const float* Wo = (const float*)d_in[11];  const float* bo = (const float*)d_in[12];
  const float* ln1g = (const float*)d_in[13]; const float* ln1b = (const float*)d_in[14];
  const float* W1 = (const float*)d_in[15];  const float* b1 = (const float*)d_in[16];
  const float* W2 = (const float*)d_in[17];  const float* b2 = (const float*)d_in[18];
  const float* ln2g = (const float*)d_in[19]; const float* ln2b = (const float*)d_in[20];
  const float* lnfg = (const float*)d_in[21]; const float* lnfb = (const float*)d_in[22];

  char* base = (char*)d_ws;
  size_t off = 0;
  auto take = [&](size_t bytes) -> void* {
    void* p = base + off;
    off += (bytes + 255) & ~(size_t)255;
    return p;
  };
  float*    x    = (float*)take((size_t)NPAD * DMODEL * 4);
  float*    y    = (float*)take((size_t)NPAD * DMODEL * 4);
  _Float16* xh   = (_Float16*)take((size_t)NPAD * DMODEL * 2);
  _Float16* qb_  = (_Float16*)take((size_t)NPAD * DMODEL * 2);
  _Float16* kb_  = (_Float16*)take((size_t)NPAD * DMODEL * 2);
  _Float16* ctx  = (_Float16*)take((size_t)NPAD * DMODEL * 2);
  _Float16* h1   = (_Float16*)take((size_t)NPAD * FFDIM * 2);
  _Float16* vtb  = (_Float16*)take((size_t)BATCH * DMODEL * SPAD * 2);
  const size_t WSQ = (size_t)DMODEL * DMODEL;  // 262144
  const size_t WSF = (size_t)DMODEL * FFDIM;   // 1048576
  const size_t LSTR = 4 * WSQ + 2 * WSF;
  _Float16* wtb  = (_Float16*)take(NLAYER * LSTR * 2);

  const dim3 blk(256);

  // ---- prep: zero V-transpose pad, convert+transpose all weights to f16
  {
    const int nvt = BATCH * DMODEL * SPAD;
    zero_h_kernel<<<(nvt + 255) / 256, blk, 0, stream>>>(vtb, nvt);
    for (int i = 0; i < NLAYER; ++i) {
      _Float16* wt = wtb + (size_t)i * LSTR;
      transposeW_kernel<<<WSQ / 256, blk, 0, stream>>>(Wq + i * WSQ, wt + 0 * WSQ, DMODEL, 9);
      transposeW_kernel<<<WSQ / 256, blk, 0, stream>>>(Wk + i * WSQ, wt + 1 * WSQ, DMODEL, 9);
      transposeW_kernel<<<WSQ / 256, blk, 0, stream>>>(Wv + i * WSQ, wt + 2 * WSQ, DMODEL, 9);
      transposeW_kernel<<<WSQ / 256, blk, 0, stream>>>(Wo + i * WSQ, wt + 3 * WSQ, DMODEL, 9);
      transposeW_kernel<<<WSF / 256, blk, 0, stream>>>(W1 + i * WSF, wt + 4 * WSQ, DMODEL, 11);
      transposeW_kernel<<<WSF / 256, blk, 0, stream>>>(W2 + i * WSF, wt + 4 * WSQ + WSF, FFDIM, 9);
    }
  }

  embed_kernel<<<(NT * DMODEL) / 256, blk, 0, stream>>>(traj, W_emb, b_emb, pe,
                                                        x, xh);

  const dim3 gD(DMODEL / 128, NT / 128);  // (4, 200)
  const dim3 gF(FFDIM / 128, NT / 128);   // (16, 200)
  const dim3 gA(2, NHEAD, BATCH);

  for (int i = 0; i < NLAYER; ++i) {
    const size_t d1 = (size_t)i * DMODEL;
    _Float16* wt = wtb + (size_t)i * LSTR;
    gemm_kernel<false, false, true, false><<<gD, blk, 0, stream>>>(
        xh, wt + 0 * WSQ, bq + d1, nullptr, qb_, NT, DMODEL, DMODEL);
    gemm_kernel<false, false, true, false><<<gD, blk, 0, stream>>>(
        xh, wt + 1 * WSQ, bk + d1, nullptr, kb_, NT, DMODEL, DMODEL);
    gemm_kernel<false, false, false, true><<<gD, blk, 0, stream>>>(
        xh, wt + 2 * WSQ, bv + d1, nullptr, vtb, NT, DMODEL, DMODEL);
    attn_kernel<<<gA, blk, 0, stream>>>(qb_, kb_, vtb, mask, ctx);
    gemm_kernel<false, true, false, false><<<gD, blk, 0, stream>>>(
        ctx, wt + 3 * WSQ, bo + d1, y, nullptr, NT, DMODEL, DMODEL);
    ln_kernel<true, true, true><<<NT / 8, blk, 0, stream>>>(
        x, y, ln1g + d1, ln1b + d1, x, xh, NT);
    gemm_kernel<true, false, true, false><<<gF, blk, 0, stream>>>(
        xh, wt + 4 * WSQ, b1 + (size_t)i * FFDIM, nullptr, h1, NT, FFDIM,
        DMODEL);
    gemm_kernel<false, true, false, false><<<gD, blk, 0, stream>>>(
        h1, wt + 4 * WSQ + WSF, b2 + d1, y, nullptr, NT, DMODEL, FFDIM);
    ln_kernel<true, true, true><<<NT / 8, blk, 0, stream>>>(
        x, y, ln2g + d1, ln2b + d1, x, xh, NT);
  }
  ln_kernel<false, true, false><<<NT / 8, blk, 0, stream>>>(
      x, nullptr, lnfg, lnfb, (float*)d_out, nullptr, NT);

  (void)in_sizes; (void)n_in; (void)out_size; (void)ws_size;
}
